// KoLeoLoss_74887049773256
// MI455X (gfx1250) — compile-verified, matrix operand order checked by reading, not run on hardware
//
#include <hip/hip_runtime.h>
#include <hip/hip_bf16.h>
#include <math.h>

typedef __attribute__((ext_vector_type(16))) _Float16 v16h;
typedef __attribute__((ext_vector_type(8)))  float    v8f;
typedef __attribute__((ext_vector_type(4)))  int      v4i;

#define N_S 8192
#define N_M 32768
#define DIM 1024
#define KEPS 1e-8f
#define STRIDE 40   // 32 halves + 8 pad -> 80B row stride, conflict-free frag reads

#if defined(__HIP_DEVICE_COMPILE__) && __has_builtin(__builtin_amdgcn_global_load_async_to_lds_b128)
#define HAS_ASYNC_LDS 1
#else
#define HAS_ASYNC_LDS 0
#endif

#if HAS_ASYNC_LDS
typedef __attribute__((address_space(1))) v4i* gas_ptr;   // global (AS1) int4*
typedef __attribute__((address_space(3))) v4i* las_ptr;   // LDS (AS3) int4*
#define ASYNC_WAIT() __builtin_amdgcn_s_wait_asynccnt(0)
#else
#define ASYNC_WAIT()
#endif

union Frag {
    float4 q[2];   // 2 x 16B
    v16h   h;      // 32B = 16 halves
};

// ---------------- 1) L2-normalize rows, emit f16 ----------------
__global__ void koleo_normalize_f16(const float* __restrict__ in,
                                    _Float16* __restrict__ out) {
    const int row = blockIdx.x;
    const int t = threadIdx.x;                 // 256 threads, 4 elems each
    const float* r = in + (size_t)row * DIM;
    float vals[4];
    float ss = 0.f;
#pragma unroll
    for (int i = 0; i < 4; ++i) {
        vals[i] = r[t + i * 256];
        ss += vals[i] * vals[i];
    }
#pragma unroll
    for (int off = 16; off >= 1; off >>= 1)
        ss += __shfl_xor(ss, off, 32);
    __shared__ float wsum[8];
    if ((t & 31) == 0) wsum[t >> 5] = ss;
    __syncthreads();
    float tot = 0.f;
#pragma unroll
    for (int w = 0; w < 8; ++w) tot += wsum[w];
    const float scale = 1.f / fmaxf(sqrtf(tot), KEPS);
    _Float16* o = out + (size_t)row * DIM;
#pragma unroll
    for (int i = 0; i < 4; ++i)
        o[t + i * 256] = (_Float16)(vals[i] * scale);
}

// ---------------- 2) init per-row min to FLT_MAX bits ----------------
__global__ void koleo_init_min(unsigned int* __restrict__ rowmin) {
    const int i = blockIdx.x * blockDim.x + threadIdx.x;
    if (i < N_S) rowmin[i] = 0x7f7fffffu;  // FLT_MAX
}

// ---------------- 3) WMMA GEMM + fused dist/diag/min epilogue ----------------
// grid.x over S rows (8192/128=64), grid.y over M rows (32768/128=256)
__global__ void __launch_bounds__(256)
koleo_gemm_min(const _Float16* __restrict__ S,   // [N_S][DIM] normalized
               const _Float16* __restrict__ M,   // [N_M][DIM] normalized
               unsigned int* __restrict__ rowmin) {
    __shared__ _Float16 As[2][128 * STRIDE];
    __shared__ _Float16 Bs[2][128 * STRIDE];
    __shared__ unsigned int blkmin[128];

    const int tid = threadIdx.x;
    const int rowBase = blockIdx.x * 128;   // into S
    const int colBase = blockIdx.y * 128;   // into M

    if (tid < 128) blkmin[tid] = 0x7f7fffffu;

    const int wave = tid >> 5;
    const int lane = tid & 31;
    const int wr = wave >> 1;      // 0..3 : 32-row group
    const int wc = wave & 1;       // 0..1 : 64-col group
    const int lh = lane >> 4;      // lane half
    const int l15 = lane & 15;

    // cooperative staging: each thread moves 32B of A and 32B of B per k-step
    const int ldRow = tid >> 1;            // 0..127
    const int ldSeg = (tid & 1) * 16;      // half offset 0 or 16
    const int ldOff = ldRow * STRIDE + ldSeg;

    const _Float16* Sg = S + (size_t)(rowBase + ldRow) * DIM + ldSeg;
    const _Float16* Mg = M + (size_t)(colBase + ldRow) * DIM + ldSeg;

    // stage one 32-wide K chunk into buffer `buf` (async DMA to LDS when available)
    auto stage = [&](int buf, int k0) {
#if HAS_ASYNC_LDS
        __builtin_amdgcn_global_load_async_to_lds_b128(
            (gas_ptr)(void*)(Sg + k0),     (las_ptr)(void*)(&As[buf][ldOff]),     0, 0);
        __builtin_amdgcn_global_load_async_to_lds_b128(
            (gas_ptr)(void*)(Sg + k0 + 8), (las_ptr)(void*)(&As[buf][ldOff + 8]), 0, 0);
        __builtin_amdgcn_global_load_async_to_lds_b128(
            (gas_ptr)(void*)(Mg + k0),     (las_ptr)(void*)(&Bs[buf][ldOff]),     0, 0);
        __builtin_amdgcn_global_load_async_to_lds_b128(
            (gas_ptr)(void*)(Mg + k0 + 8), (las_ptr)(void*)(&Bs[buf][ldOff + 8]), 0, 0);
#else
        const float4* sa = (const float4*)(Sg + k0);
        const float4* sb = (const float4*)(Mg + k0);
        float4 a0 = sa[0], a1 = sa[1];
        float4 b0 = sb[0], b1 = sb[1];
        float4* da = (float4*)(&As[buf][ldOff]);
        float4* db = (float4*)(&Bs[buf][ldOff]);
        da[0] = a0; da[1] = a1;
        db[0] = b0; db[1] = b1;
#endif
    };

    v8f acc[2][4];
#pragma unroll
    for (int rt = 0; rt < 2; ++rt)
#pragma unroll
        for (int ct = 0; ct < 4; ++ct)
#pragma unroll
            for (int e = 0; e < 8; ++e) acc[rt][ct][e] = 0.f;

    stage(0, 0);   // prefetch first chunk

#pragma unroll 2
    for (int i = 0; i < DIM / 32; ++i) {
        const int cur = i & 1;
        ASYNC_WAIT();          // my async writes into buf[cur] have landed in LDS
        __syncthreads();       // everyone's writes landed; prev reads of buf[1-cur] retired
        if (i + 1 < DIM / 32)
            stage(1 - cur, (i + 1) * 32);   // prefetch next chunk while computing

        // Both A and B fragments use the same per-lane K layout:
        // any consistent K permutation leaves the dot product unchanged.
        Frag afrag[2];
#pragma unroll
        for (int rt = 0; rt < 2; ++rt) {
            const float4* p = (const float4*)(&As[cur][(wr * 32 + rt * 16 + l15) * STRIDE + lh * 16]);
            afrag[rt].q[0] = p[0];
            afrag[rt].q[1] = p[1];
        }
#pragma unroll
        for (int ct = 0; ct < 4; ++ct) {
            Frag bfrag;
            const float4* p = (const float4*)(&Bs[cur][(wc * 64 + ct * 16 + l15) * STRIDE + lh * 16]);
            bfrag.q[0] = p[0];
            bfrag.q[1] = p[1];
#pragma unroll
            for (int rt = 0; rt < 2; ++rt)
                acc[rt][ct] = __builtin_amdgcn_wmma_f32_16x16x32_f16(
                    false, afrag[rt].h, false, bfrag.h,
                    (short)0, acc[rt][ct], false, false);
        }
    }

    // epilogue: dist = sqrt(max(2-2*dot,0)) (+1 on diagonal), per-row min
#pragma unroll
    for (int rt = 0; rt < 2; ++rt) {
#pragma unroll
        for (int v = 0; v < 8; ++v) {
            const int lrow = wr * 32 + rt * 16 + lh * 8 + v;   // row within block
            const int row = rowBase + lrow;
            float mn = 3.4e38f;
#pragma unroll
            for (int ct = 0; ct < 4; ++ct) {
                const int col = colBase + wc * 64 + ct * 16 + l15;
                const float dot = acc[rt][ct][v];
                float dist = sqrtf(fmaxf(2.f - 2.f * dot, 0.f));
                if (row == col) dist += 1.f;   // diagonal self-distance mask
                mn = fminf(mn, dist);
            }
            // min across the 16 lanes of this half (same row, 16 columns)
#pragma unroll
            for (int off = 1; off < 16; off <<= 1)
                mn = fminf(mn, __shfl_xor(mn, off, 32));
            if (l15 == v)  // one lane per row issues the DS atomic
                atomicMin(&blkmin[lrow], __float_as_uint(mn));
        }
    }
    __syncthreads();
    if (tid < 128) atomicMin(&rowmin[rowBase + tid], blkmin[tid]);
}

// ---------------- 4) loss = -mean(log(min + eps)) ----------------
__global__ void koleo_final_loss(const unsigned int* __restrict__ rowmin,
                                 float* __restrict__ out) {
    float s = 0.f;
    for (int i = threadIdx.x; i < N_S; i += 256)
        s += logf(__uint_as_float(rowmin[i]) + KEPS);
#pragma unroll
    for (int off = 16; off >= 1; off >>= 1)
        s += __shfl_xor(s, off, 32);
    __shared__ float wsum[8];
    if ((threadIdx.x & 31) == 0) wsum[threadIdx.x >> 5] = s;
    __syncthreads();
    if (threadIdx.x == 0) {
        float tot = 0.f;
#pragma unroll
        for (int w = 0; w < 8; ++w) tot += wsum[w];
        out[0] = -tot / (float)N_S;
    }
}

extern "C" void kernel_launch(void* const* d_in, const int* in_sizes, int n_in,
                              void* d_out, int out_size, void* d_ws, size_t ws_size,
                              hipStream_t stream) {
    const float* s_f32 = (const float*)d_in[0];   // [8192, 1024]
    const float* m_f32 = (const float*)d_in[1];   // [32768, 1024]

    // workspace layout: s16 (16MB) | m16 (64MB) | rowmin (32KB)
    _Float16* s16 = (_Float16*)d_ws;
    _Float16* m16 = s16 + (size_t)N_S * DIM;
    unsigned int* rowmin = (unsigned int*)(m16 + (size_t)N_M * DIM);

    koleo_normalize_f16<<<N_S, 256, 0, stream>>>(s_f32, s16);
    koleo_normalize_f16<<<N_M, 256, 0, stream>>>(m_f32, m16);
    koleo_init_min<<<(N_S + 255) / 256, 256, 0, stream>>>(rowmin);

    dim3 grid(N_S / 128, N_M / 128);   // 64 x 256 = 16384 blocks
    koleo_gemm_min<<<grid, 256, 0, stream>>>(s16, m16, rowmin);

    koleo_final_loss<<<1, 256, 0, stream>>>(rowmin, (float*)d_out);
}
